// ResidualAttentionBlock_MoA_15023795601992
// MI455X (gfx1250) — matrix operand order repaired and checked by reading, not compiled, not run on hardware
//
#include <hip/hip_runtime.h>
#include <cstdint>

// ---------------------------------------------------------------------------
// CDNA5 (gfx1250) wave32 WMMA types
// ---------------------------------------------------------------------------
typedef __attribute__((ext_vector_type(16))) __bf16 v16bf;
typedef __attribute__((ext_vector_type(8)))  float  v8f;

union ABu { unsigned int u[8]; v16bf v; };

__device__ __forceinline__ v8f wmma_bf16(v16bf a, v16bf b, v8f c) {
  return __builtin_amdgcn_wmma_f32_16x16x32_bf16(false, a, false, b, (short)0, c,
                                                 false, false);
}

__device__ __forceinline__ v8f zero8() {
  v8f z = {0.f, 0.f, 0.f, 0.f, 0.f, 0.f, 0.f, 0.f};
  return z;
}

// A fragment, 16x32 bf16 (M x K), row-major source, p = &A[m_row][k0].
// Contiguous k order used here: lane l holds row m=l%16; VGPR i: k = 2i + 16*(l>=16).
__device__ __forceinline__ v16bf load_frag_a(const __bf16* p, int lane) {
  const unsigned kb = (lane >= 16) ? 16u : 0u;
  ABu r;
  const unsigned int* q0 = (const unsigned int*)(p + kb);
#pragma unroll
  for (int i = 0; i < 8; ++i) r.u[i] = q0[i];
  return r.v;
}

// B fragment, 32x16 bf16 (K x N) with B[k][n] = W[n][k], W row-major [N,K],
// p = &W[n_col][k0]. Lane l holds col n=l%16; VGPR i: k = 2i + 16*(l>=16).
__device__ __forceinline__ v16bf load_frag_b(const __bf16* p, int lane) {
  const unsigned kb = (lane >= 16) ? 16u : 0u;
  ABu r;
  const unsigned int* q0 = (const unsigned int*)(p + kb);
#pragma unroll
  for (int i = 0; i < 8; ++i) r.u[i] = q0[i];
  return r.v;
}

// ---------------------------------------------------------------------------
// Problem constants
// ---------------------------------------------------------------------------
constexpr int Lq = 197, Nb = 128, Dm = 768, Hh = 12, Ex = 8, Ff = 64;
constexpr int Mrows = Lq * Nb;   // 25216 = 64 * 394
constexpr int D3 = 3 * Dm;       // 2304

// ---------------------------------------------------------------------------
// fp32 -> bf16 conversion (weights)
// ---------------------------------------------------------------------------
__global__ void cvt_kernel(const float* __restrict__ in, __bf16* __restrict__ out, int n) {
  int i = blockIdx.x * blockDim.x + threadIdx.x;
  int stride = gridDim.x * blockDim.x;
  for (; i < n; i += stride) out[i] = (__bf16)in[i];
}

// ---------------------------------------------------------------------------
// LayerNorm: one wave per 768-wide row, output bf16
// ---------------------------------------------------------------------------
__global__ void __launch_bounds__(256)
ln_kernel(const float* __restrict__ x, const float* __restrict__ g,
          const float* __restrict__ b, __bf16* __restrict__ out, int M) {
  const int lane = threadIdx.x & 31;
  const int row  = blockIdx.x * (blockDim.x >> 5) + (threadIdx.x >> 5);
  if (row >= M) return;
  const float* xr = x + (size_t)row * Dm;
  float v[24];
  float s = 0.f;
#pragma unroll
  for (int i = 0; i < 24; ++i) { v[i] = xr[lane + 32 * i]; s += v[i]; }
#pragma unroll
  for (int off = 16; off >= 1; off >>= 1) s += __shfl_xor(s, off);
  const float mean = s * (1.f / 768.f);
  float var = 0.f;
#pragma unroll
  for (int i = 0; i < 24; ++i) { float d = v[i] - mean; var += d * d; }
#pragma unroll
  for (int off = 16; off >= 1; off >>= 1) var += __shfl_xor(var, off);
  const float rstd = rsqrtf(var * (1.f / 768.f) + 1e-5f);
  __bf16* o = out + (size_t)row * Dm;
#pragma unroll
  for (int i = 0; i < 24; ++i) {
    int c = lane + 32 * i;
    o[c] = (__bf16)((v[i] - mean) * rstd * g[c] + b[c]);
  }
}

// ---------------------------------------------------------------------------
// Epilogue helper shared by both GEMM kernels.
// EPI: 0 = bf16 store, 1 = +r0 residual -> fp32 o32 + bf16 o16,
//      2 = QuickGELU -> bf16, 3 = +r0 +r1 -> fp32 (final), 4 = ReLU -> bf16
// ---------------------------------------------------------------------------
template <int EPI>
__device__ __forceinline__ void epilogue_tile(
    v8f acc, int m0, int n0, int lane, int N,
    const float* __restrict__ bias,
    const float* __restrict__ r0, const float* __restrict__ r1,
    float* __restrict__ o32, __bf16* __restrict__ o16) {
  const int col = n0 + (lane & 15);
  const int rb  = m0 + ((lane >= 16) ? 8 : 0);
  const float bv = bias[col];
#pragma unroll
  for (int r = 0; r < 8; ++r) {
    const int row = rb + r;
    const size_t idx = (size_t)row * N + col;
    float c = acc[r] + bv;
    if (EPI == 0) {
      o16[idx] = (__bf16)c;
    } else if (EPI == 1) {
      float vv = c + r0[idx];
      o32[idx] = vv;
      o16[idx] = (__bf16)vv;
    } else if (EPI == 2) {
      float gg = c / (1.f + __expf(-1.702f * c));   // QuickGELU
      o16[idx] = (__bf16)gg;
    } else if (EPI == 3) {
      o32[idx] = c + r0[idx] + r1[idx];
    } else {  // EPI == 4, ReLU
      o16[idx] = (__bf16)(c > 0.f ? c : 0.f);
    }
  }
}

// ---------------------------------------------------------------------------
// Fat GEMM: each wave computes a 32x64 register tile (2 A-frags x 4 B-frags
// -> 8 WMMAs per k-step from 12 b128 loads). Block = 8 waves as 2(M) x 4(N)
// -> 64x256 block tile. Requires M % 64 == 0 and N % 256 == 0.
// ---------------------------------------------------------------------------
template <int EPI>
__global__ void __launch_bounds__(256)
gemm2_kernel(const __bf16* __restrict__ A, const __bf16* __restrict__ W,
             const float* __restrict__ bias,
             const float* __restrict__ r0, const float* __restrict__ r1,
             float* __restrict__ o32, __bf16* __restrict__ o16,
             int M, int N, int K) {
  const int lane  = threadIdx.x & 31;
  const int wave  = threadIdx.x >> 5;
  const int waveM = wave >> 2;            // 0..1
  const int waveN = wave & 3;             // 0..3
  const int m0 = blockIdx.y * 64 + waveM * 32;
  const int n0 = blockIdx.x * 256 + waveN * 64;

  const __bf16* arow0 = A + (size_t)(m0 + (lane & 15)) * K;
  const __bf16* arow1 = arow0 + (size_t)16 * K;
  const __bf16* wrow  = W + (size_t)(n0 + (lane & 15)) * K;
  const size_t wstep  = (size_t)16 * K;

  v8f acc[2][4];
#pragma unroll
  for (int i = 0; i < 2; ++i)
#pragma unroll
    for (int j = 0; j < 4; ++j) acc[i][j] = zero8();

  for (int k0 = 0; k0 < K; k0 += 32) {
    const v16bf a0 = load_frag_a(arow0 + k0, lane);
    const v16bf a1 = load_frag_a(arow1 + k0, lane);
    v16bf bf[4];
#pragma unroll
    for (int j = 0; j < 4; ++j) bf[j] = load_frag_b(wrow + j * wstep + k0, lane);
#pragma unroll
    for (int j = 0; j < 4; ++j) {
      acc[0][j] = wmma_bf16(a0, bf[j], acc[0][j]);
      acc[1][j] = wmma_bf16(a1, bf[j], acc[1][j]);
    }
  }

#pragma unroll
  for (int i = 0; i < 2; ++i)
#pragma unroll
    for (int j = 0; j < 4; ++j)
      epilogue_tile<EPI>(acc[i][j], m0 + i * 16, n0 + j * 16, lane, N,
                         bias, r0, r1, o32, o16);
}

// ---------------------------------------------------------------------------
// Simple GEMM (one 16x16 tile per wave) — used for narrow N (adapter down).
// ---------------------------------------------------------------------------
template <int EPI>
__global__ void __launch_bounds__(256)
gemm_kernel(const __bf16* __restrict__ A, const __bf16* __restrict__ W,
            const float* __restrict__ bias,
            const float* __restrict__ r0, const float* __restrict__ r1,
            float* __restrict__ o32, __bf16* __restrict__ o16,
            int M, int N, int K) {
  const int lane = threadIdx.x & 31;
  const int wave = threadIdx.x >> 5;
  const int n0 = (blockIdx.x * (blockDim.x >> 5) + wave) * 16;
  const int m0 = blockIdx.y * 16;
  if (n0 >= N || m0 >= M) return;

  const __bf16* arow = A + (size_t)(m0 + (lane & 15)) * K;
  const __bf16* wrow = W + (size_t)(n0 + (lane & 15)) * K;

  v8f acc = zero8();
  for (int k0 = 0; k0 < K; k0 += 32) {
    v16bf a = load_frag_a(arow + k0, lane);
    v16bf b = load_frag_b(wrow + k0, lane);
    acc = wmma_bf16(a, b, acc);
  }
  epilogue_tile<EPI>(acc, m0, n0, lane, N, bias, r0, r1, o32, o16);
}

// ---------------------------------------------------------------------------
// Attention: one block per (n,h); 8 waves; wave handles 16-query tiles
// t = wave, wave+8 (13 tiles cover L=197, padded rows masked at store).
// Online (flash-style) softmax over 32-key super-tiles.
// ---------------------------------------------------------------------------
__global__ void __launch_bounds__(256)
attn_kernel(const __bf16* __restrict__ qkv, __bf16* __restrict__ ao) {
  const int n = blockIdx.x / Hh;
  const int h = blockIdx.x % Hh;
  const int lane = threadIdx.x & 31;
  const int wave = threadIdx.x >> 5;
  const int l16 = lane & 15;
  const int half16 = (lane >= 16) ? 1 : 0;

  __shared__ __bf16 plds[8][16 * 32];   // per-wave P bounce buffer (8 KB)
  __bf16* pb = &plds[wave][0];

  const __bf16* qbase = qkv + h * 64;
  const __bf16* kbase = qkv + Dm + h * 64;
  const __bf16* vbase = qkv + 2 * Dm + h * 64;

  for (int t = wave; t < 13; t += 8) {
    const int r0 = t * 16;
    int qm = r0 + l16; if (qm > Lq - 1) qm = Lq - 1;       // clamp pad rows
    const __bf16* qrow = qbase + (size_t)(qm * Nb + n) * D3;
    const v16bf qa0 = load_frag_a(qrow, lane);
    const v16bf qa1 = load_frag_a(qrow + 32, lane);

    float rmax[8], rsum[8];
    v8f o[4];
#pragma unroll
    for (int r = 0; r < 8; ++r) { rmax[r] = -1e30f; rsum[r] = 0.f; }
#pragma unroll
    for (int dt = 0; dt < 4; ++dt) o[dt] = zero8();

    for (int jj = 0; jj < 7; ++jj) {        // 7*32 = 224 >= 197 keys
      const int c0 = jj * 32;
      v8f s0 = zero8(), s1 = zero8();
#pragma unroll
      for (int hh = 0; hh < 2; ++hh) {
        const int kc = c0 + hh * 16 + l16;
        const int kcl = kc > Lq - 1 ? Lq - 1 : kc;
        const __bf16* krow = kbase + (size_t)(kcl * Nb + n) * D3;
        v8f sv = zero8();
        sv = wmma_bf16(qa0, load_frag_b(krow, lane), sv);
        sv = wmma_bf16(qa1, load_frag_b(krow + 32, lane), sv);
        const bool msk = (kc > Lq - 1);
#pragma unroll
        for (int r = 0; r < 8; ++r) sv[r] = msk ? -1e30f : sv[r] * 0.125f;
        if (hh == 0) s0 = sv; else s1 = sv;
      }
      // online softmax row stats (rows confined to 16-lane halves)
      float scl[8];
#pragma unroll
      for (int r = 0; r < 8; ++r) {
        float mx = fmaxf(s0[r], s1[r]);
#pragma unroll
        for (int off = 1; off < 16; off <<= 1) mx = fmaxf(mx, __shfl_xor(mx, off));
        const float nm = fmaxf(rmax[r], mx);
        scl[r] = __expf(rmax[r] - nm);
        rmax[r] = nm;
        const float p0 = __expf(s0[r] - nm);
        const float p1 = __expf(s1[r] - nm);
        s0[r] = p0; s1[r] = p1;
        float ts = p0 + p1;
#pragma unroll
        for (int off = 1; off < 16; off <<= 1) ts += __shfl_xor(ts, off);
        rsum[r] = rsum[r] * scl[r] + ts;
      }
#pragma unroll
      for (int dt = 0; dt < 4; ++dt)
#pragma unroll
        for (int r = 0; r < 8; ++r) o[dt][r] *= scl[r];

      // C-fragment P -> LDS -> A-fragment re-layout
#pragma unroll
      for (int r = 0; r < 8; ++r) {
        const int pr = half16 * 8 + r;
        pb[pr * 32 + l16]      = (__bf16)s0[r];
        pb[pr * 32 + 16 + l16] = (__bf16)s1[r];
      }
      asm volatile("s_wait_dscnt 0" ::: "memory");
      const v16bf pa = load_frag_a(pb + l16 * 32, lane);

      // O += P @ V  (4 column tiles of 16 over head-dim 64)
#pragma unroll
      for (int dt = 0; dt < 4; ++dt) {
        ABu vb;
        const int d = dt * 16 + l16;
        const int kb2 = half16 * 16;
#pragma unroll
        for (int i = 0; i < 8; ++i) {
          int lv = c0 + kb2 + 2 * i;
          const int ca = lv > Lq - 1 ? Lq - 1 : lv;
          const int cb = (lv + 1) > Lq - 1 ? Lq - 1 : (lv + 1);
          const unsigned a0 = *(const unsigned short*)(vbase + (size_t)(ca * Nb + n) * D3 + d);
          const unsigned a1 = *(const unsigned short*)(vbase + (size_t)(cb * Nb + n) * D3 + d);
          vb.u[i] = a0 | (a1 << 16);
        }
        o[dt] = wmma_bf16(pa, vb.v, o[dt]);
      }
    }

    // normalize + store (skip padded query rows)
    const int rb = r0 + half16 * 8;
#pragma unroll
    for (int dt = 0; dt < 4; ++dt)
#pragma unroll
      for (int r = 0; r < 8; ++r) {
        const int row = rb + r;
        if (row < Lq)
          ao[(size_t)(row * Nb + n) * Dm + h * 64 + dt * 16 + l16] =
              (__bf16)(o[dt][r] / rsum[r]);
      }
  }
}

// ---------------------------------------------------------------------------
// Router + noisy-top-2 gating (eval: no noise) on CLS rows of x1
// ---------------------------------------------------------------------------
__global__ void router_kernel(const float* __restrict__ x1,
                              const float* __restrict__ router,
                              float* __restrict__ gates) {
  const int nn = blockIdx.x * blockDim.x + threadIdx.x;
  if (nn >= Nb) return;
  const float* row = x1 + (size_t)nn * Dm;   // l=0 -> row index = n
  float lg[Ex];
#pragma unroll
  for (int e = 0; e < Ex; ++e) lg[e] = 0.f;
  for (int d = 0; d < Dm; ++d) {
    const float xv = row[d];
#pragma unroll
    for (int e = 0; e < Ex; ++e) lg[e] += xv * router[d * Ex + e];
  }
  int i0 = 0;
#pragma unroll
  for (int e = 1; e < Ex; ++e) if (lg[e] > lg[i0]) i0 = e;
  int i1 = (i0 == 0) ? 1 : 0;
#pragma unroll
  for (int e = 0; e < Ex; ++e) if (e != i0 && lg[e] > lg[i1]) i1 = e;
  const float m  = lg[i0];
  const float e0 = 1.f;                       // exp(0)
  const float e1 = __expf(lg[i1] - m);
  const float inv = 1.f / (e0 + e1);
#pragma unroll
  for (int e = 0; e < Ex; ++e) gates[nn * Ex + e] = 0.f;
  gates[nn * Ex + i0] = e0 * inv;
  gates[nn * Ex + i1] = e1 * inv;
}

// ---------------------------------------------------------------------------
// Adapter up: y = 0.1 * ( sum_e (g_e .* h_e) @ up_w[e]^T + sum_e g_e * up_b[e] )
// Gate folded into the A fragment rows before WMMA.
// ---------------------------------------------------------------------------
__global__ void __launch_bounds__(256)
adapter_up_kernel(const __bf16* __restrict__ h, const __bf16* __restrict__ upw,
                  const float* __restrict__ upb, const float* __restrict__ gates,
                  float* __restrict__ y, int M) {
  const int lane = threadIdx.x & 31;
  const int wave = threadIdx.x >> 5;
  const int n0 = (blockIdx.x * (blockDim.x >> 5) + wave) * 16;
  const int m0 = blockIdx.y * 16;
  const int ma = m0 + (lane & 15);

  v8f acc = zero8();
#pragma unroll
  for (int e = 0; e < Ex; ++e) {
    const float g = gates[(ma & (Nb - 1)) * Ex + e];
    const __bf16* he = h + ((size_t)e * M + ma) * Ff;
    const __bf16* we = upw + (size_t)e * Dm * Ff + (size_t)(n0 + (lane & 15)) * Ff;
#pragma unroll
    for (int ks = 0; ks < 2; ++ks) {
      v16bf a = load_frag_a(he + ks * 32, lane);
#pragma unroll
      for (int i = 0; i < 16; ++i) a[i] = (__bf16)((float)a[i] * g);
      v16bf b = load_frag_b(we + ks * 32, lane);
      acc = wmma_bf16(a, b, acc);
    }
  }

  const int col = n0 + (lane & 15);
  const int rb  = m0 + ((lane >= 16) ? 8 : 0);
#pragma unroll
  for (int r = 0; r < 8; ++r) {
    const int row = rb + r;
    const int nbk = row & (Nb - 1);
    float bs = 0.f;
#pragma unroll
    for (int e = 0; e < Ex; ++e) bs += gates[nbk * Ex + e] * upb[e * Dm + col];
    y[(size_t)row * Dm + col] = 0.1f * (acc[r] + bs);
  }
}

// ---------------------------------------------------------------------------
// Host launch
// ---------------------------------------------------------------------------
extern "C" void kernel_launch(void* const* d_in, const int* in_sizes, int n_in,
                              void* d_out, int out_size, void* d_ws, size_t ws_size,
                              hipStream_t stream) {
  (void)in_sizes; (void)n_in; (void)out_size; (void)ws_size;
  const float* x      = (const float*)d_in[0];
  const float* ln1_g  = (const float*)d_in[1];
  const float* ln1_b  = (const float*)d_in[2];
  const float* qkv_w  = (const float*)d_in[3];
  const float* qkv_b  = (const float*)d_in[4];
  const float* out_w  = (const float*)d_in[5];
  const float* out_b  = (const float*)d_in[6];
  const float* ln2_g  = (const float*)d_in[7];
  const float* ln2_b  = (const float*)d_in[8];
  const float* fc_w   = (const float*)d_in[9];
  const float* fc_b   = (const float*)d_in[10];
  const float* proj_w = (const float*)d_in[11];
  const float* proj_b = (const float*)d_in[12];
  const float* routerw= (const float*)d_in[13];
  const float* down_w = (const float*)d_in[14];
  const float* down_b = (const float*)d_in[15];
  const float* up_w   = (const float*)d_in[16];
  const float* up_b   = (const float*)d_in[17];
  float* out = (float*)d_out;

  const int M = Mrows;
  char* wsb = (char*)d_ws;
  size_t off = 0;
  auto take = [&](size_t bytes) -> char* {
    char* p = wsb + off;
    off += (bytes + 255) & ~(size_t)255;
    return p;
  };
  __bf16* xn    = (__bf16*)take((size_t)M * Dm * 2);     // LN1/LN2 output (reused)
  __bf16* qkvb  = (__bf16*)take((size_t)M * D3 * 2);
  __bf16* aob   = (__bf16*)take((size_t)M * Dm * 2);
  float*  x1    = (float*) take((size_t)M * Dm * 4);
  __bf16* x1b   = (__bf16*)take((size_t)M * Dm * 2);
  float*  gates = (float*) take((size_t)Nb * Ex * 4);
  __bf16* hexp  = (__bf16*)take((size_t)Ex * M * Ff * 2);
  float*  yad   = (float*) take((size_t)M * Dm * 4);
  __bf16* hmb   = (__bf16*)take((size_t)M * 4 * Dm * 2);
  __bf16* wq    = (__bf16*)take((size_t)D3 * Dm * 2);
  __bf16* wo    = (__bf16*)take((size_t)Dm * Dm * 2);
  __bf16* wfc   = (__bf16*)take((size_t)4 * Dm * Dm * 2);
  __bf16* wpj   = (__bf16*)take((size_t)Dm * 4 * Dm * 2);
  __bf16* wdn   = (__bf16*)take((size_t)Ex * Ff * Dm * 2);
  __bf16* wup   = (__bf16*)take((size_t)Ex * Dm * Ff * 2);

  auto cvt = [&](const float* src, __bf16* dst, int n) {
    int blocks = (n + 255) / 256; if (blocks > 2048) blocks = 2048;
    cvt_kernel<<<blocks, 256, 0, stream>>>(src, dst, n);
  };
  cvt(qkv_w, wq, D3 * Dm);
  cvt(out_w, wo, Dm * Dm);
  cvt(fc_w, wfc, 4 * Dm * Dm);
  cvt(proj_w, wpj, Dm * 4 * Dm);
  cvt(down_w, wdn, Ex * Ff * Dm);
  cvt(up_w, wup, Ex * Dm * Ff);

  // 1. LN1
  ln_kernel<<<M / 8, 256, 0, stream>>>(x, ln1_g, ln1_b, xn, M);
  // 2. QKV GEMM -> bf16 [M, 2304]   (64x256 block tiles)
  gemm2_kernel<0><<<dim3(D3 / 256, M / 64), 256, 0, stream>>>(
      xn, wq, qkv_b, nullptr, nullptr, nullptr, qkvb, M, D3, Dm);
  // 3. Attention -> ao bf16 [M, 768]
  attn_kernel<<<Nb * Hh, 256, 0, stream>>>(qkvb, aob);
  // 4. out-proj + residual -> x1 fp32 + bf16
  gemm2_kernel<1><<<dim3(Dm / 256, M / 64), 256, 0, stream>>>(
      aob, wo, out_b, x, nullptr, x1, x1b, M, Dm, Dm);
  // 5. Router + top-2 gates
  router_kernel<<<1, 128, 0, stream>>>(x1, routerw, gates);
  // 6. Adapter down (per expert): h_e = relu(x1 @ down_w[e]^T + down_b[e])
  for (int e = 0; e < Ex; ++e) {
    gemm_kernel<4><<<dim3(1, M / 16), 128, 0, stream>>>(
        x1b, wdn + (size_t)e * Ff * Dm, down_b + e * Ff,
        nullptr, nullptr, nullptr, hexp + (size_t)e * M * Ff, M, Ff, Dm);
  }
  // 7. Adapter up, gate-weighted combine -> y fp32
  adapter_up_kernel<<<dim3(Dm / 128, M / 16), 256, 0, stream>>>(
      hexp, wup, up_b, gates, yad, M);
  // 8. LN2 (reuse xn buffer)
  ln_kernel<<<M / 8, 256, 0, stream>>>(x1, ln2_g, ln2_b, xn, M);
  // 9. FC + QuickGELU -> hm bf16 [M, 3072]
  gemm2_kernel<2><<<dim3(4 * Dm / 256, M / 64), 256, 0, stream>>>(
      xn, wfc, fc_b, nullptr, nullptr, nullptr, hmb, M, 4 * Dm, Dm);
  // 10. proj + x1 + y -> out fp32
  gemm2_kernel<3><<<dim3(Dm / 256, M / 64), 256, 0, stream>>>(
      hmb, wpj, proj_b, x1, yad, out, nullptr, M, Dm, 4 * Dm);
}
